// MultiHeadDiffAttention_16492674416938
// MI455X (gfx1250) — matrix operand order repaired and checked by reading, not compiled, not run on hardware
//
#include <hip/hip_runtime.h>
#include <math.h>

// ---------------- constants ----------------
#define C_EMBD   1024
#define NHEAD    16
#define HEADD    64
#define TSEQ     1024
#define NBATCH   4
#define ROWS     4096            // B*T
#define LAMBDA_INIT_F  0.35550906759096926f
#define ONE_MINUS_LI   0.6444909324090307f

// ---------------- CDNA5 WMMA types ----------------
typedef __attribute__((ext_vector_type(16))) __bf16 bf16x16;
typedef __attribute__((ext_vector_type(8)))  float  f32x8;
typedef __attribute__((ext_vector_type(4)))  unsigned int uint32x4;
typedef __attribute__((ext_vector_type(8)))  int    int32x8;
typedef __attribute__((ext_vector_type(4)))  int    int32x4;

__device__ __forceinline__ f32x8 wmma_bf16(bf16x16 a, bf16x16 b, f32x8 c) {
    return __builtin_amdgcn_wmma_f32_16x16x32_bf16(false, a, false, b,
                                                   (short)0, c, false, false);
}

// A-fragment 16x32 bf16 from row-major src (rows = M, ld in bf16 units).
__device__ __forceinline__ bf16x16 load_frag_a(const __bf16* src, int ld) {
    int lane = threadIdx.x & 31;
    int m  = lane & 15;
    int kh = lane >> 4;
    union { bf16x16 f; uint4 u[2]; } r;
    r.u[0] = *reinterpret_cast<const uint4*>(src + (size_t)m * ld + kh * 8);
    r.u[1] = *reinterpret_cast<const uint4*>(src + (size_t)m * ld + 16 + kh * 8);
    return r.f;
}

// B-fragment 32x16 bf16; src row-major with rows = N, ld in bf16 units.
__device__ __forceinline__ bf16x16 load_frag_b(const __bf16* src, int ld) {
    int lane = threadIdx.x & 31;
    int n  = lane & 15;
    int kh = lane >> 4;
    union { bf16x16 f; uint4 u[2]; } r;
    r.u[0] = *reinterpret_cast<const uint4*>(src + (size_t)n * ld + kh * 8);
    r.u[1] = *reinterpret_cast<const uint4*>(src + (size_t)n * ld + 16 + kh * 8);
    return r.f;
}

__device__ __forceinline__ f32x8 zero8() {
    f32x8 z;
#pragma unroll
    for (int i = 0; i < 8; ++i) z[i] = 0.0f;
    return z;
}

// ---------------- Tensor Data Mover: 2D bf16 tile global -> LDS ----------------
// Builds a D# per CDNA5 ISA 8.3/8.4 and issues TENSOR_LOAD_TO_LDS.
// tile_w elems/row, tile_h rows, src row stride = stride_elems (bf16 units).
// LDS rows padded by pad-amount DWORDs after each pad-interval DWORDs:
//   pad_int_code: 0=2,1=4,2=8,3=16,4=32,... DWORDs   pad_amt_code: 0=1,1=2,... DWORDs
__device__ __forceinline__ void tdm_load_2d(const __bf16* gsrc, __bf16* ldst,
                                            int tile_w, int tile_h, int stride_elems,
                                            int pad_int_code, int pad_amt_code) {
    unsigned long long ga = (unsigned long long)(uintptr_t)gsrc;
    unsigned int       la = (unsigned int)(uintptr_t)ldst;   // LDS addr = low 32 bits

    uint32x4 g0;
    g0[0] = 1u;                                              // count=1 (user descriptor)
    g0[1] = la;                                              // lds_addr
    g0[2] = (unsigned int)(ga & 0xFFFFFFFFu);                // global_addr[31:0]
    g0[3] = (unsigned int)((ga >> 32) & 0x01FFFFFFu) | 0x80000000u; // addr[56:32] | type=2

    unsigned int td0 = (unsigned int)stride_elems;           // tensor_dim0 (row length)
    unsigned int td1 = (unsigned int)tile_h;                 // tensor_dim1
    unsigned int dw0 = (1u << 16);                           // data_size=1 (2 bytes)
    dw0 |= (1u << 20) | ((unsigned int)pad_int_code << 22) | ((unsigned int)pad_amt_code << 25);

    int32x8 g1;
    g1[0] = (int)dw0;
    g1[1] = (int)((td0 & 0xFFFFu) << 16);                    // abar=0 | tensor_dim0 lo
    g1[2] = (int)(((td0 >> 16) & 0xFFFFu) | ((td1 & 0xFFFFu) << 16));
    g1[3] = (int)(((td1 >> 16) & 0xFFFFu) | ((unsigned int)tile_w << 16)); // tile_dim0
    g1[4] = (int)((unsigned int)tile_h & 0xFFFFu);           // tile_dim1 (tile_dim2=0)
    g1[5] = (int)(unsigned int)stride_elems;                 // tensor_dim0_stride lo32
    g1[6] = 0;
    g1[7] = 0;

    int32x4 gz; gz[0] = 0; gz[1] = 0; gz[2] = 0; gz[3] = 0;
#if __clang_major__ >= 23
    int32x8 gz8; 
#pragma unroll
    for (int i = 0; i < 8; ++i) gz8[i] = 0;
    __builtin_amdgcn_tensor_load_to_lds(g0, g1, gz, gz, gz8, 0);
#else
    __builtin_amdgcn_tensor_load_to_lds(g0, g1, gz, gz, 0);
#endif
}

// ---------------- BatchNorm statistics ----------------
__global__ __launch_bounds__(256) void bn_stats(const float* __restrict__ x,
                                                const float* __restrict__ gamma,
                                                const float* __restrict__ beta,
                                                float* __restrict__ scale,
                                                float* __restrict__ shift) {
    const int c = blockIdx.x;
    float s = 0.f, s2 = 0.f;
    for (int r = threadIdx.x; r < ROWS; r += 256) {
        float v = x[(size_t)r * C_EMBD + c];
        s += v; s2 += v * v;
    }
    __shared__ float rs[256], rq[256];
    rs[threadIdx.x] = s; rq[threadIdx.x] = s2;
    __syncthreads();
    for (int off = 128; off > 0; off >>= 1) {
        if ((int)threadIdx.x < off) {
            rs[threadIdx.x] += rs[threadIdx.x + off];
            rq[threadIdx.x] += rq[threadIdx.x + off];
        }
        __syncthreads();
    }
    if (threadIdx.x == 0) {
        float mean = rs[0] * (1.0f / ROWS);
        float var  = rq[0] * (1.0f / ROWS) - mean * mean;
        float sc   = gamma[c] * rsqrtf(var + 1e-5f);
        scale[c] = sc;
        shift[c] = beta[c] - mean * sc;
    }
}

__global__ __launch_bounds__(256) void bn_apply(const float* __restrict__ x,
                                                const float* __restrict__ scale,
                                                const float* __restrict__ shift,
                                                __bf16* __restrict__ xn, int n) {
    int i = blockIdx.x * 256 + threadIdx.x;
    if (i < n) {
        int c = i & (C_EMBD - 1);
        xn[i] = (__bf16)(x[i] * scale[c] + shift[c]);
    }
}

__global__ __launch_bounds__(256) void f32_to_bf16_k(const float* __restrict__ s,
                                                     __bf16* __restrict__ d, int n) {
    int i = blockIdx.x * 256 + threadIdx.x;
    if (i < n) d[i] = (__bf16)s[i];
}

__global__ void lambda_kernel(const float* __restrict__ lq1, const float* __restrict__ lk1,
                              const float* __restrict__ lq2, const float* __restrict__ lk2,
                              float* __restrict__ lam) {
    int i = threadIdx.x;   // 32 threads, 64 elements
    float a = lq1[i] * lk1[i] + lq1[i + 32] * lk1[i + 32];
    float b = lq2[i] * lk2[i] + lq2[i + 32] * lk2[i + 32];
#pragma unroll
    for (int m = 16; m >= 1; m >>= 1) {
        a += __shfl_xor(a, m, 32);
        b += __shfl_xor(b, m, 32);
    }
    if (i == 0) *lam = __expf(a) - __expf(b) + LAMBDA_INIT_F;
}

// ---------------- NT GEMM: C[M,N] = A[M,K] * B[N,K]^T (both row-major bf16) ---
// TDM double-buffered tiles: issue k+1 while computing k, synced via TENSORcnt.
template <bool OUT_BF16>
__global__ __launch_bounds__(256) void gemm_nt(const __bf16* __restrict__ A,
                                               const __bf16* __restrict__ B,
                                               float* __restrict__ Cf,
                                               __bf16* __restrict__ Cb,
                                               int M, int N, int K) {
    __shared__ __bf16 As[2][128 * 40];
    __shared__ __bf16 Bs[2][128 * 40];

    const int tid  = threadIdx.x;
    const int wave = tid >> 5;
    const int wm   = wave >> 2;       // 0..1
    const int wn   = wave & 3;        // 0..3
    const int bm   = blockIdx.y * 128;
    const int bn   = blockIdx.x * 128;

    f32x8 acc[4][2];
#pragma unroll
    for (int i = 0; i < 4; ++i)
#pragma unroll
        for (int j = 0; j < 2; ++j) acc[i][j] = zero8();

    // prologue: tile 0 into buffer 0 (wave 0 drives the TDM)
    if (tid < 32) {
        tdm_load_2d(A + (size_t)bm * K, &As[0][0], 32, 128, K, 3, 3);
        tdm_load_2d(B + (size_t)bn * K, &Bs[0][0], 32, 128, K, 3, 3);
    }

    int buf = 0;
    for (int k0 = 0; k0 < K; k0 += 32, buf ^= 1) {
        const bool more = (k0 + 32) < K;
        if (tid < 32) {
            if (more) {
                tdm_load_2d(A + (size_t)bm * K + k0 + 32, &As[buf ^ 1][0], 32, 128, K, 3, 3);
                tdm_load_2d(B + (size_t)bn * K + k0 + 32, &Bs[buf ^ 1][0], 32, 128, K, 3, 3);
                __builtin_amdgcn_s_wait_tensorcnt(2);   // current buffer landed
            } else {
                __builtin_amdgcn_s_wait_tensorcnt(0);
            }
        }
        __syncthreads();

        bf16x16 af[4], bfr[2];
#pragma unroll
        for (int i = 0; i < 4; ++i) af[i]  = load_frag_a(&As[buf][(wm * 64 + i * 16) * 40], 40);
#pragma unroll
        for (int j = 0; j < 2; ++j) bfr[j] = load_frag_b(&Bs[buf][(wn * 32 + j * 16) * 40], 40);
#pragma unroll
        for (int i = 0; i < 4; ++i)
#pragma unroll
            for (int j = 0; j < 2; ++j)
                acc[i][j] = wmma_bf16(af[i], bfr[j], acc[i][j]);
        __syncthreads();   // all reads of this buffer done before it is refilled
    }

    const int lane = tid & 31;
    const int cn   = lane & 15;
    const int hh   = lane >> 4;
#pragma unroll
    for (int i = 0; i < 4; ++i)
#pragma unroll
        for (int j = 0; j < 2; ++j)
#pragma unroll
            for (int r = 0; r < 8; ++r) {
                int row = bm + wm * 64 + i * 16 + r + 8 * hh;
                int col = bn + wn * 32 + j * 16 + cn;
                float v = acc[i][j][r];
                if (OUT_BF16) Cb[(size_t)row * N + col] = (__bf16)v;
                else          Cf[(size_t)row * N + col] = v;
            }
}

// ---------------- Single-branch flash attention ----------------
__global__ __launch_bounds__(256) void attn_branch(const __bf16* __restrict__ Q,
                                                   const __bf16* __restrict__ Kg,
                                                   const __bf16* __restrict__ Vg,
                                                   __bf16* __restrict__ O) {
    __shared__ __bf16 Ks[32 * 72];
    __shared__ __bf16 Vs[128 * 40];       // transposed: [dim][key]
    __shared__ __bf16 Ps[8][16 * 40];     // per-wave P scratch (16x32)

    const int bh   = blockIdx.y;
    const int bidx = bh >> 4;
    const int h    = bh & 15;
    const int q0   = blockIdx.x * 128;
    const int tid  = threadIdx.x;
    const int wave = tid >> 5;
    const int lane = tid & 31;
    const int ln   = lane & 15;
    const int hh   = lane >> 4;

    const int qrow = q0 + wave * 16;
    const size_t qbase = (size_t)(bidx * TSEQ + qrow) * C_EMBD + h * HEADD;

    bf16x16 qa[2];
#pragma unroll
    for (int dh = 0; dh < 2; ++dh)
        qa[dh] = load_frag_a(Q + qbase + dh * 32, C_EMBD);

    float mi[8], li[8];
    f32x8 oa[8];
#pragma unroll
    for (int i = 0; i < 8; ++i) { mi[i] = -1e30f; li[i] = 0.f; oa[i] = zero8(); }

    const float sc = 0.125f;   // 1/sqrt(64)

    for (int kc = 0; kc < TSEQ; kc += 32) {
        __syncthreads();
        // K tile 32x64 via TDM (row 128B -> pad_interval code 4, pad 16B -> code 3)
        if (tid < 32)
            tdm_load_2d(Kg + (size_t)(bidx * TSEQ + kc) * C_EMBD + h * HEADD,
                        &Ks[0], 64, 32, C_EMBD, 4, 3);
        {   // V tile transposed into Vs[dim][key] (manual: TDM cannot transpose)
            int r  = tid >> 3;
            int c0 = (tid & 7) * 16;
            union { uint4 u[2]; __bf16 b[16]; } vv;
            const uint4* vp = reinterpret_cast<const uint4*>(
                Vg + (size_t)(bidx * TSEQ + kc + r) * (2 * C_EMBD) + h * 128 + c0);
            vv.u[0] = vp[0]; vv.u[1] = vp[1];
#pragma unroll
            for (int e = 0; e < 16; ++e) Vs[(c0 + e) * 40 + r] = vv.b[e];
        }
        if (tid < 32) __builtin_amdgcn_s_wait_tensorcnt(0);
        __syncthreads();

        // S = Q * K^T  (16 queries x 32 keys)
        f32x8 s[2];
        s[0] = zero8(); s[1] = zero8();
#pragma unroll
        for (int nt = 0; nt < 2; ++nt)
#pragma unroll
            for (int dh = 0; dh < 2; ++dh)
                s[nt] = wmma_bf16(qa[dh], load_frag_b(&Ks[(nt * 16) * 72 + dh * 32], 72), s[nt]);

        // online softmax
#pragma unroll
        for (int i = 0; i < 8; ++i) {
            float v0 = s[0][i] * sc, v1 = s[1][i] * sc;
            float mx = fmaxf(v0, v1);
#pragma unroll
            for (int m = 8; m >= 1; m >>= 1) mx = fmaxf(mx, __shfl_xor(mx, m, 32));
            float mnew  = fmaxf(mi[i], mx);
            float alpha = __expf(mi[i] - mnew);
            float p0 = __expf(v0 - mnew), p1 = __expf(v1 - mnew);
            float ps = p0 + p1;
#pragma unroll
            for (int m = 8; m >= 1; m >>= 1) ps += __shfl_xor(ps, m, 32);
            li[i] = li[i] * alpha + ps;
            mi[i] = mnew;
            int prow = i + 8 * hh;
            Ps[wave][prow * 40 + ln]      = (__bf16)p0;
            Ps[wave][prow * 40 + 16 + ln] = (__bf16)p1;
#pragma unroll
            for (int f = 0; f < 8; ++f) oa[f][i] *= alpha;
        }
        {
            bf16x16 pa = load_frag_a(&Ps[wave][0], 40);
#pragma unroll
            for (int f = 0; f < 8; ++f)
                oa[f] = wmma_bf16(pa, load_frag_b(&Vs[(f * 16) * 40], 40), oa[f]);
        }
    }

    // epilogue: normalize by row-sum, store bf16
#pragma unroll
    for (int i = 0; i < 8; ++i) {
        float inv = 1.0f / li[i];
        int t = q0 + wave * 16 + i + 8 * hh;
        __bf16* yp = O + (size_t)(bidx * TSEQ + t) * (2 * C_EMBD) + h * 128;
#pragma unroll
        for (int f = 0; f < 8; ++f) {
            int col = f * 16 + ln;
            yp[col] = (__bf16)(oa[f][i] * inv);
        }
    }
}

// ---------------- combine: y = y1 - lam*y2, RMSNorm(128), subln, (1-li) ------
__global__ __launch_bounds__(256) void combine_rms(const __bf16* __restrict__ Y1,
                                                   const __bf16* __restrict__ Y2,
                                                   const float* __restrict__ lam_p,
                                                   const float* __restrict__ subln,
                                                   __bf16* __restrict__ Y) {
    const int gw   = blockIdx.x * 8 + (threadIdx.x >> 5);
    const int lane = threadIdx.x & 31;
    const int row  = gw >> 4;
    const int h    = gw & 15;
    const float lam = *lam_p;
    const size_t base = (size_t)row * (2 * C_EMBD) + h * 128 + lane * 4;

    union { uint2 u; __bf16 b[4]; } a1, a2;
    a1.u = *reinterpret_cast<const uint2*>(Y1 + base);
    a2.u = *reinterpret_cast<const uint2*>(Y2 + base);
    float v[4]; float ss = 0.f;
#pragma unroll
    for (int e = 0; e < 4; ++e) {
        float t = (float)a1.b[e] - lam * (float)a2.b[e];
        v[e] = t; ss += t * t;
    }
#pragma unroll
    for (int m = 16; m >= 1; m >>= 1) ss += __shfl_xor(ss, m, 32);
    float rn = rsqrtf(ss * (1.0f / 128.0f) + 1e-5f);
    union { uint2 u; __bf16 b[4]; } o;
#pragma unroll
    for (int e = 0; e < 4; ++e)
        o.b[e] = (__bf16)(v[e] * rn * subln[lane * 4 + e] * ONE_MINUS_LI);
    *reinterpret_cast<uint2*>(Y + base) = o.u;
}

// ---------------- host orchestration ----------------
extern "C" void kernel_launch(void* const* d_in, const int* in_sizes, int n_in,
                              void* d_out, int out_size, void* d_ws, size_t ws_size,
                              hipStream_t stream) {
    (void)in_sizes; (void)n_in; (void)out_size; (void)ws_size;
    const float* x     = (const float*)d_in[0];
    const float* q1w   = (const float*)d_in[1];
    const float* q2w   = (const float*)d_in[2];
    const float* k1w   = (const float*)d_in[3];
    const float* k2w   = (const float*)d_in[4];
    const float* vw    = (const float*)d_in[5];
    const float* cw    = (const float*)d_in[6];
    const float* subln = (const float*)d_in[7];
    const float* lq1   = (const float*)d_in[8];
    const float* lk1   = (const float*)d_in[9];
    const float* lq2   = (const float*)d_in[10];
    const float* lk2   = (const float*)d_in[11];
    const float* gam   = (const float*)d_in[12];
    const float* bet   = (const float*)d_in[13];
    float* out = (float*)d_out;

    char* ws = (char*)d_ws;
    size_t cur = 0;
    auto alloc = [&](size_t bytes) { char* p = ws + cur; cur = (cur + bytes + 255) & ~(size_t)255; return p; };

    float*  scale = (float*)alloc(C_EMBD * 4);
    float*  shift = (float*)alloc(C_EMBD * 4);
    float*  lam   = (float*)alloc(256);
    __bf16* xn    = (__bf16*)alloc((size_t)ROWS * C_EMBD * 2);
    __bf16* wq1   = (__bf16*)alloc((size_t)C_EMBD * C_EMBD * 2);
    __bf16* wq2   = (__bf16*)alloc((size_t)C_EMBD * C_EMBD * 2);
    __bf16* wk1   = (__bf16*)alloc((size_t)C_EMBD * C_EMBD * 2);
    __bf16* wk2   = (__bf16*)alloc((size_t)C_EMBD * C_EMBD * 2);
    __bf16* wv    = (__bf16*)alloc((size_t)2 * C_EMBD * C_EMBD * 2);
    __bf16* wc    = (__bf16*)alloc((size_t)C_EMBD * 2 * C_EMBD * 2);
    __bf16* q1    = (__bf16*)alloc((size_t)ROWS * C_EMBD * 2);
    __bf16* q2    = (__bf16*)alloc((size_t)ROWS * C_EMBD * 2);
    __bf16* k1    = (__bf16*)alloc((size_t)ROWS * C_EMBD * 2);
    __bf16* k2    = (__bf16*)alloc((size_t)ROWS * C_EMBD * 2);
    __bf16* v     = (__bf16*)alloc((size_t)ROWS * 2 * C_EMBD * 2);
    __bf16* y1    = (__bf16*)alloc((size_t)ROWS * 2 * C_EMBD * 2);
    __bf16* y2    = (__bf16*)alloc((size_t)ROWS * 2 * C_EMBD * 2);
    __bf16* y     = (__bf16*)alloc((size_t)ROWS * 2 * C_EMBD * 2);

    // 1) BN stats + apply
    bn_stats<<<C_EMBD, 256, 0, stream>>>(x, gam, bet, scale, shift);
    bn_apply<<<(ROWS * C_EMBD) / 256, 256, 0, stream>>>(x, scale, shift, xn, ROWS * C_EMBD);

    // 2) weight conversion + lambda
    const int n1 = C_EMBD * C_EMBD, n2 = 2 * C_EMBD * C_EMBD;
    f32_to_bf16_k<<<n1 / 256, 256, 0, stream>>>(q1w, wq1, n1);
    f32_to_bf16_k<<<n1 / 256, 256, 0, stream>>>(q2w, wq2, n1);
    f32_to_bf16_k<<<n1 / 256, 256, 0, stream>>>(k1w, wk1, n1);
    f32_to_bf16_k<<<n1 / 256, 256, 0, stream>>>(k2w, wk2, n1);
    f32_to_bf16_k<<<n2 / 256, 256, 0, stream>>>(vw, wv, n2);
    f32_to_bf16_k<<<n2 / 256, 256, 0, stream>>>(cw, wc, n2);
    lambda_kernel<<<1, 32, 0, stream>>>(lq1, lk1, lq2, lk2, lam);

    // 3) projections: [4096,K] x [N,K]^T
    dim3 blk(256);
    dim3 gq(C_EMBD / 128, ROWS / 128);            // (8,32)
    gemm_nt<true><<<gq, blk, 0, stream>>>(xn, wq1, nullptr, q1, ROWS, C_EMBD, C_EMBD);
    gemm_nt<true><<<gq, blk, 0, stream>>>(xn, wq2, nullptr, q2, ROWS, C_EMBD, C_EMBD);
    gemm_nt<true><<<gq, blk, 0, stream>>>(xn, wk1, nullptr, k1, ROWS, C_EMBD, C_EMBD);
    gemm_nt<true><<<gq, blk, 0, stream>>>(xn, wk2, nullptr, k2, ROWS, C_EMBD, C_EMBD);
    dim3 gv(2 * C_EMBD / 128, ROWS / 128);        // (16,32)
    gemm_nt<true><<<gv, blk, 0, stream>>>(xn, wv, nullptr, v, ROWS, 2 * C_EMBD, C_EMBD);

    // 4) flash attention, one branch per launch
    dim3 ga(TSEQ / 128, NBATCH * NHEAD);          // (8,64)
    attn_branch<<<ga, blk, 0, stream>>>(q1, k1, v, y1);
    attn_branch<<<ga, blk, 0, stream>>>(q2, k2, v, y2);

    // 5) combine + RMSNorm + scales
    combine_rms<<<(ROWS * NHEAD) / 8, 256, 0, stream>>>(y1, y2, lam, subln, y);

    // 6) output projection: [4096,2048] x [1024,2048]^T -> f32 out
    dim3 go(C_EMBD / 128, ROWS / 128);            // (8,32)
    gemm_nt<false><<<go, blk, 0, stream>>>(y, wc, out, nullptr, ROWS, C_EMBD, 2 * C_EMBD);
}
